// MultiHeadedAttention_180388627337
// MI455X (gfx1250) — matrix operand order repaired and checked by reading, not compiled, not run on hardware
//
#include <hip/hip_runtime.h>

typedef __attribute__((ext_vector_type(16))) _Float16 v16h;
typedef __attribute__((ext_vector_type(8)))  _Float16 v8h;
typedef __attribute__((ext_vector_type(8)))  float    v8f;

#define FRAME_NUM 64
#define JOINT_NUM 25
#define SEQ       1600
#define INPUT_DIM 256
#define H_NUM     8
#define H_DIM     64
#define DMODEL    512
#define BATCH     8
#define NEGF      -3.0e38f   // -> -inf in f16; exp() == 0

#define CAT16(lo, hi) __builtin_shufflevector((lo), (hi), 0, 1, 2, 3, 4, 5, 6, 7, \
                                              8, 9, 10, 11, 12, 13, 14, 15)

// ---------------------------------------------------------------------------
// Kernel 0: one-shot precision/layout prep.
//   xh[b*S+s][k]          = f16(x)                       (3,276,800 elems)
//   wt[wsel][n][k]        = f16(W[k][n])  (transposed)   (  393,216 elems)
// Makes every WMMA fragment downstream a contiguous vector load.
// ---------------------------------------------------------------------------
__global__ __launch_bounds__(256)
void cvt_kernel(const float* __restrict__ x,
                const float* __restrict__ wq, const float* __restrict__ wk,
                const float* __restrict__ wv,
                _Float16* __restrict__ xh, _Float16* __restrict__ wt)
{
    const size_t NX = (size_t)BATCH * SEQ * INPUT_DIM;
    size_t tid = (size_t)blockIdx.x * 256 + threadIdx.x;
    if (tid < NX) {
        xh[tid] = (_Float16)x[tid];
        return;
    }
    size_t i = tid - NX;
    if (i >= (size_t)3 * DMODEL * INPUT_DIM) return;
    int wsel = (int)(i >> 17);               // / (512*256)
    int rem  = (int)(i & (DMODEL * INPUT_DIM - 1));
    int n    = rem >> 8;                     // / 256
    int k    = rem & 255;
    const float* w = (wsel == 0) ? wq : (wsel == 1) ? wk : wv;
    wt[i] = (_Float16)w[(size_t)k * DMODEL + n];
}

// ---------------------------------------------------------------------------
// Kernel 1: fused QKV projection, all-f16 operands, f32 accumulate.
// 4 waves/block; each wave owns one 16x16 tile (K=256 => 8 wmma).
// Q,K stored [B][H][S][64]; V stored TRANSPOSED [B][H][64][S] (+ReLU).
// ---------------------------------------------------------------------------
__global__ __launch_bounds__(128)
void qkv_proj_kernel(const _Float16* __restrict__ xh,
                     const _Float16* __restrict__ wt,
                     const float* __restrict__ bq, const float* __restrict__ bk,
                     const float* __restrict__ bv,
                     _Float16* __restrict__ qf, _Float16* __restrict__ kf,
                     _Float16* __restrict__ vt)
{
    const int lane  = threadIdx.x & 31;
    const int wave  = threadIdx.x >> 5;
    const int mtile = blockIdx.x;            // 0..799
    const int n96   = blockIdx.y * 4 + wave; // 0..95
    const int wsel  = n96 >> 5;              // 0:Q 1:K 2:V
    const int ntile = n96 & 31;

    const int half = lane >> 4;
    const int ml   = lane & 15;
    const int m0   = mtile << 4;
    const int ncol = (ntile << 4) + ml;

    // A row pointer (f16 x); B column pointer (f16 transposed W, contiguous K)
    const _Float16* xrow = xh + (size_t)(m0 + ml) * INPUT_DIM + 16 * half;
    const _Float16* wrow = wt + ((size_t)wsel * DMODEL + ncol) * INPUT_DIM + 16 * half;

    v8f c = {};
#pragma unroll
    for (int kt = 0; kt < INPUT_DIM; kt += 32) {
        v8h  alo = *(const v8h*)(xrow + kt);        // K = kt+16h .. +7
        v8h  ahi = *(const v8h*)(xrow + kt + 8);    // K = kt+16h+8 .. +15
        v16h a   = CAT16(alo, ahi);
        v16h bfr = *(const v16h*)(wrow + kt);       // K = kt+16h .. +15
        c = __builtin_amdgcn_wmma_f32_16x16x32_f16(false, a, false, bfr,
                                                   (short)0, c, false, false);
    }

    const float* bias = (wsel == 0) ? bq : (wsel == 1) ? bk : bv;
    const float  bb   = bias[ncol];
    const int    h    = ncol >> 6;
    const int    d    = ncol & 63;
    const int    bidx = mtile / (SEQ / 16);          // uniform scalar div
    const int    sbase = (mtile - bidx * (SEQ / 16)) * 16;

    if (wsel == 2) {
        // V transposed: [B][H][64][S]; lane's 8 results are contiguous in s.
        _Float16* dstp = vt + (((size_t)(bidx * H_NUM + h) * H_DIM + d) * SEQ)
                            + sbase + 8 * half;
        v8h o;
#pragma unroll
        for (int r = 0; r < 8; ++r) {
            float v = c[r] + bb;
            o[r] = (_Float16)(v > 0.f ? v : 0.f);
        }
        *(v8h*)dstp = o;
    } else {
        _Float16* dst  = (wsel == 0) ? qf : kf;
        _Float16* dstp = dst + ((size_t)(bidx * H_NUM + h) * SEQ + sbase + 8 * half)
                             * H_DIM + d;
#pragma unroll
        for (int r = 0; r < 8; ++r)
            dstp[(size_t)r * H_DIM] = (_Float16)(c[r] + bb);
    }
}

// ---------------------------------------------------------------------------
// Kernel 2: masked attention, one block per (b, h, 16-query tile).
// 4 waves: QK^T wmma + mask -> f16 LDS; cooperative softmax (fp32 p_attn out,
// single expf pass); P@V wmma from LDS (transposed V => contiguous B frags).
// ---------------------------------------------------------------------------
__global__ __launch_bounds__(128)
void attn_kernel(const _Float16* __restrict__ qf,
                 const _Float16* __restrict__ kf,
                 const _Float16* __restrict__ vt,
                 float* __restrict__ out, float* __restrict__ pout)
{
    __shared__ _Float16 sc[16 * SEQ];        // 51200 B
    __shared__ float    red[16 * 8];
    __shared__ float    rowmax[16];
    __shared__ float    rowsum[16];

    const int tid  = threadIdx.x;
    const int lane = tid & 31;
    const int wave = tid >> 5;
    const int half = lane >> 4;
    const int ml   = lane & 15;

    const int qt = blockIdx.x;               // 0..99
    const int h  = blockIdx.y;
    const int b  = blockIdx.z;
    const int bh = b * H_NUM + h;
    const int q0 = qt << 4;

    const _Float16* qbase = qf + (size_t)bh * SEQ * H_DIM;
    const _Float16* kbase = kf + (size_t)bh * SEQ * H_DIM;
    const _Float16* vbase = vt + (size_t)bh * H_DIM * SEQ;

    // Q fragments: 16x64 as two 16x32 A fragments (contiguous b128 loads)
    v16h a0, a1;
    {
        const _Float16* qrow = qbase + (size_t)(q0 + ml) * H_DIM + 16 * half;
        v8h q00 = *(const v8h*)(qrow);
        v8h q01 = *(const v8h*)(qrow + 8);
        v8h q10 = *(const v8h*)(qrow + 32);
        v8h q11 = *(const v8h*)(qrow + 40);
        a0 = CAT16(q00, q01);
        a1 = CAT16(q10, q11);
    }

    // hoist per-row query index / frame id (kills per-tile div chains)
    int qi_r[8], qfr_r[8];
#pragma unroll
    for (int r = 0; r < 8; ++r) {
        qi_r[r]  = q0 + r + 8 * half;
        qfr_r[r] = qi_r[r] / JOINT_NUM;
    }

    // ---- scores = (Q K^T)/8 with temporal mask ----
    for (int t = wave; t < SEQ / 16; t += 4) {       // 25 tiles per wave
        const _Float16* krow = kbase + (size_t)(t * 16 + ml) * H_DIM + 16 * half;
        v16h b0 = *(const v16h*)(krow);              // d = 16h .. 16h+15
        v16h b1 = *(const v16h*)(krow + 32);
        v8f c = {};
        c = __builtin_amdgcn_wmma_f32_16x16x32_f16(false, a0, false, b0,
                                                   (short)0, c, false, false);
        c = __builtin_amdgcn_wmma_f32_16x16x32_f16(false, a1, false, b1,
                                                   (short)0, c, false, false);
        const int kj  = t * 16 + ml;
        const int kfr = kj / JOINT_NUM;              // one div per tile
#pragma unroll
        for (int r = 0; r < 8; ++r) {
            float s = c[r] * 0.125f;                 // 1/sqrt(64)
            if ((kfr == qfr_r[r]) && (kj != qi_r[r])) s = NEGF;
            sc[(r + 8 * half) * SEQ + kj] = (_Float16)s;
        }
    }
    __syncthreads();

    // ---- softmax: row = tid&15, 8 segments x 200 keys ----
    const int row = tid & 15;
    const int seg = tid >> 4;
    const int c0  = seg * (SEQ / 8);
    _Float16* srow = &sc[row * SEQ + c0];

    float mx = -3.4e38f;
    for (int j = 0; j < SEQ / 8; j += 8) {
        v8h hv = *(const v8h*)(srow + j);
#pragma unroll
        for (int e = 0; e < 8; ++e) mx = fmaxf(mx, (float)hv[e]);
    }
    red[row * 8 + seg] = mx;
    __syncthreads();
    if (seg == 0) {
        float m2 = red[row * 8];
#pragma unroll
        for (int j2 = 1; j2 < 8; ++j2) m2 = fmaxf(m2, red[row * 8 + j2]);
        rowmax[row] = m2;
    }
    __syncthreads();
    const float rmx = rowmax[row];

    float sum = 0.f;
    for (int j = 0; j < SEQ / 8; j += 8) {           // exp once, keep in LDS
        v8h hv = *(const v8h*)(srow + j);
        v8h e8;
#pragma unroll
        for (int e = 0; e < 8; ++e) {
            float ev = __expf((float)hv[e] - rmx);
            sum += ev;
            e8[e] = (_Float16)ev;
        }
        *(v8h*)(srow + j) = e8;
    }
    red[row * 8 + seg] = sum;
    __syncthreads();
    if (seg == 0) {
        float s2 = 0.f;
#pragma unroll
        for (int j2 = 0; j2 < 8; ++j2) s2 += red[row * 8 + j2];
        rowsum[row] = s2;
    }
    __syncthreads();

    const float inv  = 1.0f / rowsum[row];
    float*      prow = pout + ((size_t)bh * SEQ + (q0 + row)) * SEQ + c0;
    for (int j = 0; j < SEQ / 8; j += 8) {
        v8h e8 = *(const v8h*)(srow + j);
        v8h p8;
        float4 f0, f1;
        float pv[8];
#pragma unroll
        for (int e = 0; e < 8; ++e) {
            pv[e] = (float)e8[e] * inv;
            p8[e] = (_Float16)pv[e];
        }
        f0 = make_float4(pv[0], pv[1], pv[2], pv[3]);
        f1 = make_float4(pv[4], pv[5], pv[6], pv[7]);
        *(float4*)(prow + j)     = f0;               // fp32 p_attn (655 MB)
        *(float4*)(prow + j + 4) = f1;
        *(v8h*)(srow + j) = p8;                      // normalized f16 P
    }
    __syncthreads();

    // ---- out = P @ V : wave owns dims [wave*16, wave*16+16) ----
    const int d0 = wave << 4;
    const _Float16* vrow = vbase + (size_t)(d0 + ml) * SEQ + 16 * half;
    const _Float16* p16  = &sc[ml * SEQ + 16 * half];
    v8f cc = {};
    for (int kt = 0; kt < SEQ; kt += 32) {           // 50 wmma steps
        v8h  plo = *(const v8h*)(p16 + kt);
        v8h  phi = *(const v8h*)(p16 + kt + 8);
        v16h pa  = CAT16(plo, phi);
        v16h vb  = *(const v16h*)(vrow + kt);        // contiguous keys (V^T)
        cc = __builtin_amdgcn_wmma_f32_16x16x32_f16(false, pa, false, vb,
                                                    (short)0, cc, false, false);
    }
#pragma unroll
    for (int r = 0; r < 8; ++r) {
        int qi = q0 + r + 8 * half;
        out[((size_t)b * SEQ + qi) * DMODEL + h * H_DIM + d0 + ml] = cc[r];
    }
}

// ---------------------------------------------------------------------------
extern "C" void kernel_launch(void* const* d_in, const int* in_sizes, int n_in,
                              void* d_out, int out_size, void* d_ws, size_t ws_size,
                              hipStream_t stream) {
    const float* x  = (const float*)d_in[0];
    const float* wq = (const float*)d_in[1];
    const float* bq = (const float*)d_in[2];
    const float* wk = (const float*)d_in[3];
    const float* bk = (const float*)d_in[4];
    const float* wv = (const float*)d_in[5];
    const float* bv = (const float*)d_in[6];

    const size_t qkv_elems = (size_t)BATCH * H_NUM * SEQ * H_DIM; // 6,553,600
    const size_t x_elems   = (size_t)BATCH * SEQ * INPUT_DIM;     // 3,276,800
    const size_t w_elems   = (size_t)3 * DMODEL * INPUT_DIM;      //   393,216

    char* ws = (char*)d_ws;
    _Float16* qf = (_Float16*)(ws);
    _Float16* kf = qf + qkv_elems;
    _Float16* vt = kf + qkv_elems;       // V stored transposed [B][H][64][S]
    _Float16* xh = vt + qkv_elems;
    _Float16* wt = xh + x_elems;         // total ~46.7 MB of f16 scratch

    float* out  = (float*)d_out;                       // [8,1600,512]
    float* pout = out + (size_t)BATCH * SEQ * DMODEL;  // [8,8,1600,1600]

    const size_t cvt_total = x_elems + w_elems;        // 3,670,016
    dim3 gCvt((unsigned)((cvt_total + 255) / 256));
    cvt_kernel<<<gCvt, 256, 0, stream>>>(x, wq, wk, wv, xh, wt);

    dim3 gProj(BATCH * SEQ / 16, (3 * DMODEL / 16) / 4);   // 800 x 24, 4 waves
    qkv_proj_kernel<<<gProj, 128, 0, stream>>>(xh, wt, bq, bk, bv, qf, kf, vt);

    dim3 gAttn(SEQ / 16, H_NUM, BATCH);                    // 100 x 8 x 8
    attn_kernel<<<gAttn, 128, 0, stream>>>(qf, kf, vt, out, pout);
}
